// CrossModalBlock_10720238371152
// MI455X (gfx1250) — matrix operand order repaired and checked by reading, not compiled, not run on hardware
//
#include <hip/hip_runtime.h>
#include <hip/hip_fp16.h>

typedef __attribute__((ext_vector_type(16))) _Float16 v16h;
typedef __attribute__((ext_vector_type(8)))  _Float16 v8h;
typedef __attribute__((ext_vector_type(8)))  float    v8f;
typedef __attribute__((ext_vector_type(4)))  float    v4f;

#define DM   1024
#define NH   16
#define DH   64
#define HIDN 4096
#define BB   8
#define PP   1024
#define NN   512

// ---------- WMMA fragment loaders (wave32, 16x16x32 f16) ----------
// A 16x32 f16: lane<16 row=l, K in [kk,kk+8)+[kk+16,kk+24); lane>=16 shifted by 8.
static __device__ inline v16h load_a_f16(const _Float16* p) {
  v8h lo = *(const v8h*)(p);
  v8h hi = *(const v8h*)(p + 16);
  v16h r;
#pragma unroll
  for (int i = 0; i < 8; ++i) { r[i] = lo[i]; r[i + 8] = hi[i]; }
  return r;
}
static __device__ inline v16h load_a_f32(const float* p) {
  v4f a0 = *(const v4f*)(p);
  v4f a1 = *(const v4f*)(p + 4);
  v4f b0 = *(const v4f*)(p + 16);
  v4f b1 = *(const v4f*)(p + 20);
  v16h r;
#pragma unroll
  for (int i = 0; i < 4; ++i) {
    r[i]      = (_Float16)a0[i];
    r[i + 4]  = (_Float16)a1[i];
    r[i + 8]  = (_Float16)b0[i];
    r[i + 12] = (_Float16)b1[i];
  }
  return r;
}
// B 32x16 f16: lane<16 col=l, K in [kk,kk+8)+[kk+16,kk+24); lane>=16 col=l-16, shifted by 8.
static __device__ inline v16h load_b_f16(const _Float16* p) {
  v8h lo = *(const v8h*)(p);
  v8h hi = *(const v8h*)(p + 16);
  v16h r;
#pragma unroll
  for (int i = 0; i < 8; ++i) { r[i] = lo[i]; r[i + 8] = hi[i]; }
  return r;
}
static __device__ inline v8f wmma_step(v16h a, v16h b, v8f c) {
  return __builtin_amdgcn_wmma_f32_16x16x32_f16(false, a, false, b, (short)0, c,
                                                false, false);
}

// ---------- fp32 -> f16 convert ----------
__global__ __launch_bounds__(256) void cvt_k(const float* __restrict__ s,
                                             _Float16* __restrict__ d, int n) {
  int i = blockIdx.x * 256 + threadIdx.x;
  if (i < n) d[i] = (_Float16)s[i];
}

// ================= 32x32-per-wave GEMMs (2x2 fragments) =================

// ---------- GEMM: A fp32, B f16 (N,K row-major), out f16 (+bias, opt relu) ----------
__global__ __launch_bounds__(256) void gemm_a32_f16_k(
    const float* __restrict__ A, int lda, const _Float16* __restrict__ Bt, int ldb,
    const float* __restrict__ bias, _Float16* __restrict__ C, int ldc,
    int M, int N, int K, int relu) {
  int wid = blockIdx.x * 8 + (threadIdx.x >> 5);
  int ntx = N >> 5;
  int tiles = (M >> 5) * ntx;
  if (wid >= tiles) return;
  int tm = wid / ntx, tn = wid % ntx;
  int l = threadIdx.x & 31, la = l & 15, hq = l >> 4;
  const float*    ar0 = A  + (size_t)(tm * 32 + la) * lda + hq * 8;
  const float*    ar1 = ar0 + (size_t)16 * lda;
  const _Float16* br0 = Bt + (size_t)(tn * 32 + la) * ldb + hq * 8;
  const _Float16* br1 = br0 + (size_t)16 * ldb;
  v8f c00 = {}, c01 = {}, c10 = {}, c11 = {};
  for (int kk = 0; kk < K; kk += 32) {
    v16h a0 = load_a_f32(ar0 + kk);
    v16h a1 = load_a_f32(ar1 + kk);
    v16h b0 = load_b_f16(br0 + kk);
    v16h b1 = load_b_f16(br1 + kk);
    c00 = wmma_step(a0, b0, c00);
    c01 = wmma_step(a0, b1, c01);
    c10 = wmma_step(a1, b0, c10);
    c11 = wmma_step(a1, b1, c11);
  }
  int col0 = tn * 32 + la, col1 = col0 + 16;
  int r0 = tm * 32 + hq * 8, r1 = r0 + 16;
  float bn0 = bias[col0], bn1 = bias[col1];
#pragma unroll
  for (int r = 0; r < 8; ++r) {
    float v00 = c00[r] + bn0, v01 = c01[r] + bn1;
    float v10 = c10[r] + bn0, v11 = c11[r] + bn1;
    if (relu) {
      v00 = fmaxf(v00, 0.f); v01 = fmaxf(v01, 0.f);
      v10 = fmaxf(v10, 0.f); v11 = fmaxf(v11, 0.f);
    }
    C[(size_t)(r0 + r) * ldc + col0] = (_Float16)v00;
    C[(size_t)(r0 + r) * ldc + col1] = (_Float16)v01;
    C[(size_t)(r1 + r) * ldc + col0] = (_Float16)v10;
    C[(size_t)(r1 + r) * ldc + col1] = (_Float16)v11;
  }
}

// ---------- GEMM: A f16, B f16, out fp32 = acc + bias + residual ----------
__global__ __launch_bounds__(256) void gemm_a16_f32res_k(
    const _Float16* __restrict__ A, int lda, const _Float16* __restrict__ Bt, int ldb,
    const float* __restrict__ bias, const float* __restrict__ res,
    float* __restrict__ C, int ldc, int M, int N, int K) {
  int wid = blockIdx.x * 8 + (threadIdx.x >> 5);
  int ntx = N >> 5;
  int tiles = (M >> 5) * ntx;
  if (wid >= tiles) return;
  int tm = wid / ntx, tn = wid % ntx;
  int l = threadIdx.x & 31, la = l & 15, hq = l >> 4;
  const _Float16* ar0 = A  + (size_t)(tm * 32 + la) * lda + hq * 8;
  const _Float16* ar1 = ar0 + (size_t)16 * lda;
  const _Float16* br0 = Bt + (size_t)(tn * 32 + la) * ldb + hq * 8;
  const _Float16* br1 = br0 + (size_t)16 * ldb;
  v8f c00 = {}, c01 = {}, c10 = {}, c11 = {};
  for (int kk = 0; kk < K; kk += 32) {
    v16h a0 = load_a_f16(ar0 + kk);
    v16h a1 = load_a_f16(ar1 + kk);
    v16h b0 = load_b_f16(br0 + kk);
    v16h b1 = load_b_f16(br1 + kk);
    c00 = wmma_step(a0, b0, c00);
    c01 = wmma_step(a0, b1, c01);
    c10 = wmma_step(a1, b0, c10);
    c11 = wmma_step(a1, b1, c11);
  }
  int col0 = tn * 32 + la, col1 = col0 + 16;
  int r0 = tm * 32 + hq * 8, r1 = r0 + 16;
  float bn0 = bias[col0], bn1 = bias[col1];
#pragma unroll
  for (int r = 0; r < 8; ++r) {
    size_t i00 = (size_t)(r0 + r) * ldc + col0;
    size_t i01 = (size_t)(r0 + r) * ldc + col1;
    size_t i10 = (size_t)(r1 + r) * ldc + col0;
    size_t i11 = (size_t)(r1 + r) * ldc + col1;
    C[i00] = c00[r] + bn0 + res[i00];
    C[i01] = c01[r] + bn1 + res[i01];
    C[i10] = c10[r] + bn0 + res[i10];
    C[i11] = c11[r] + bn1 + res[i11];
  }
}

// ---------- attn_out = weights(fp32) x V^T(f16): per (b,h) M=1024 N=64 K=512 ----------
__global__ __launch_bounds__(256) void gemm_attn_k(
    const float* __restrict__ W, const _Float16* __restrict__ vt,
    _Float16* __restrict__ attn) {
  int wid = blockIdx.x * 8 + (threadIdx.x >> 5);   // 8192 waves
  int bh = wid >> 6;                               // 64 tiles of 32x32 per (b,h)
  int t = wid & 63;
  int tm = t >> 1, tn = t & 1;
  int b = bh >> 4, h = bh & 15;
  const float*    A  = W  + (size_t)bh * PP * NN;
  const _Float16* Bt = vt + (size_t)bh * DH * NN;
  int l = threadIdx.x & 31, la = l & 15, hq = l >> 4;
  const float*    ar0 = A  + (size_t)(tm * 32 + la) * NN + hq * 8;
  const float*    ar1 = ar0 + (size_t)16 * NN;
  const _Float16* br0 = Bt + (size_t)(tn * 32 + la) * NN + hq * 8;
  const _Float16* br1 = br0 + (size_t)16 * NN;
  v8f c00 = {}, c01 = {}, c10 = {}, c11 = {};
  for (int kk = 0; kk < NN; kk += 32) {
    v16h a0 = load_a_f32(ar0 + kk);
    v16h a1 = load_a_f32(ar1 + kk);
    v16h b0 = load_b_f16(br0 + kk);
    v16h b1 = load_b_f16(br1 + kk);
    c00 = wmma_step(a0, b0, c00);
    c01 = wmma_step(a0, b1, c01);
    c10 = wmma_step(a1, b0, c10);
    c11 = wmma_step(a1, b1, c11);
  }
  int col0 = tn * 32 + la, col1 = col0 + 16;
  int r0 = tm * 32 + hq * 8, r1 = r0 + 16;
#pragma unroll
  for (int r = 0; r < 8; ++r) {
    attn[(size_t)(b * PP + r0 + r) * DM + h * DH + col0] = (_Float16)c00[r];
    attn[(size_t)(b * PP + r0 + r) * DM + h * DH + col1] = (_Float16)c01[r];
    attn[(size_t)(b * PP + r1 + r) * DM + h * DH + col0] = (_Float16)c10[r];
    attn[(size_t)(b * PP + r1 + r) * DM + h * DH + col1] = (_Float16)c11[r];
  }
}

// ---------- V projection with per-head transposed store: vt[(bh*64+dd)*512+tok] ----------
__global__ __launch_bounds__(256) void gemm_vT_k(
    const float* __restrict__ A, const _Float16* __restrict__ Bt,
    const float* __restrict__ bias, _Float16* __restrict__ vt) {
  int wid = blockIdx.x * 8 + (threadIdx.x >> 5);
  const int ntx = DM >> 4;                 // 64
  int tiles = ((BB * NN) >> 4) * ntx;      // 256*64
  if (wid >= tiles) return;
  int tm = wid / ntx, tn = wid % ntx;
  int l = threadIdx.x & 31, la = l & 15, hq = l >> 4;
  const float*    ar = A  + (size_t)(tm * 16 + la) * DM + hq * 8;
  const _Float16* br = Bt + (size_t)(tn * 16 + la) * DM + hq * 8;
  v8f acc = {};
  for (int kk = 0; kk < DM; kk += 32)
    acc = wmma_step(load_a_f32(ar + kk), load_b_f16(br + kk), acc);
  int col = tn * 16 + la;                  // global out dim: h*64+dd
  int h = col >> 6, dd = col & 63;
  int r0 = tm * 16 + hq * 8;
  float bn = bias[col];
#pragma unroll
  for (int r = 0; r < 8; ++r) {
    int m = r0 + r;                        // (b, tok)
    int b = m >> 9, tok = m & 511;
    vt[((size_t)((b << 4) + h) * 64 + dd) * NN + tok] = (_Float16)(acc[r] + bn);
  }
}

// ---------- fused scores (QK^T * scale / tau) + mask + softmax -> fp32 weights ----------
__global__ __launch_bounds__(512) void scores_softmax_k(
    const _Float16* __restrict__ q16, const _Float16* __restrict__ k16,
    const unsigned char* __restrict__ mask, const float* __restrict__ log_tau,
    float* __restrict__ wout) {
  __shared__ float sc[16 * NN];            // 32 KB
  int blk = blockIdx.x;                    // b*16*64 + h*64 + pg
  int pg = blk & 63, h = (blk >> 6) & 15, b = blk >> 10;
  int w = threadIdx.x >> 5, l = threadIdx.x & 31;
  int la = l & 15, hq = l >> 4;
  int p0 = pg * 16;
  const _Float16* ar = q16 + (size_t)(b * PP + p0 + la) * DM + h * DH + hq * 8;
#pragma unroll
  for (int t = 0; t < 2; ++t) {
    int n0 = (w * 2 + t) * 16;
    const _Float16* br = k16 + (size_t)(b * NN + n0 + la) * DM + h * DH + hq * 8;
    v8f acc = {};
    acc = wmma_step(load_a_f16(ar), load_b_f16(br), acc);
    acc = wmma_step(load_a_f16(ar + 32), load_b_f16(br + 32), acc);
    int col = n0 + la, r0 = hq * 8;
#pragma unroll
    for (int r = 0; r < 8; ++r) sc[(r0 + r) * NN + col] = acc[r];
  }
  __syncthreads();
  // wave w softmaxes row w (16 elems/lane)
  float scale = 0.125f * __expf(-log_tau[h]);   // 1/sqrt(64)/tau
  float vals[16];
  float mx = -INFINITY;
#pragma unroll
  for (int i = 0; i < 16; ++i) {
    int c = l + i * 32;
    float s = sc[w * NN + c] * scale;
    if (!mask[b * NN + c]) s = -INFINITY;
    vals[i] = s;
    mx = fmaxf(mx, s);
  }
#pragma unroll
  for (int off = 16; off; off >>= 1) mx = fmaxf(mx, __shfl_xor(mx, off));
  float sum = 0.f;
#pragma unroll
  for (int i = 0; i < 16; ++i) { vals[i] = __expf(vals[i] - mx); sum += vals[i]; }
#pragma unroll
  for (int off = 16; off; off >>= 1) sum += __shfl_xor(sum, off);
  float inv = 1.f / sum;
  size_t base = ((size_t)(b * NH + h) * PP + p0 + w) * NN;
#pragma unroll
  for (int i = 0; i < 16; ++i) wout[base + l + i * 32] = vals[i] * inv;
}

// ---------- LayerNorm (one row / block, 256 thr) ----------
__global__ __launch_bounds__(256) void layernorm_k(
    const float* __restrict__ y, const float* __restrict__ g,
    const float* __restrict__ bt, float* __restrict__ x) {
  __shared__ float red[8], red2[8], bc[2];
  int row = blockIdx.x;
  const float* yr = y + (size_t)row * DM;
  float v[4], s = 0.f, s2 = 0.f;
#pragma unroll
  for (int i = 0; i < 4; ++i) {
    v[i] = yr[threadIdx.x + i * 256];
    s += v[i]; s2 += v[i] * v[i];
  }
#pragma unroll
  for (int off = 16; off; off >>= 1) { s += __shfl_xor(s, off); s2 += __shfl_xor(s2, off); }
  int w = threadIdx.x >> 5;
  if ((threadIdx.x & 31) == 0) { red[w] = s; red2[w] = s2; }
  __syncthreads();
  if (threadIdx.x == 0) {
    float S = 0.f, S2 = 0.f;
    for (int j = 0; j < 8; ++j) { S += red[j]; S2 += red2[j]; }
    float mu = S * (1.f / DM);
    float var = S2 * (1.f / DM) - mu * mu;
    bc[0] = mu; bc[1] = rsqrtf(var + 1e-5f);
  }
  __syncthreads();
  float mu = bc[0], rstd = bc[1];
#pragma unroll
  for (int i = 0; i < 4; ++i) {
    int idx = threadIdx.x + i * 256;
    x[(size_t)row * DM + idx] = (v[i] - mu) * rstd * g[idx] + bt[idx];
  }
}

// ---------- LayerNorm2 + classifier + sigmoid ----------
__global__ __launch_bounds__(256) void ln2_cls_k(
    const float* __restrict__ y, const float* __restrict__ g,
    const float* __restrict__ bt, const float* __restrict__ clsw,
    const float* __restrict__ clsb, float* __restrict__ x,
    float* __restrict__ logits, float* __restrict__ sig) {
  __shared__ float red[8], red2[8], bc[2];
  int row = blockIdx.x;
  const float* yr = y + (size_t)row * DM;
  float v[4], s = 0.f, s2 = 0.f;
#pragma unroll
  for (int i = 0; i < 4; ++i) {
    v[i] = yr[threadIdx.x + i * 256];
    s += v[i]; s2 += v[i] * v[i];
  }
#pragma unroll
  for (int off = 16; off; off >>= 1) { s += __shfl_xor(s, off); s2 += __shfl_xor(s2, off); }
  int w = threadIdx.x >> 5;
  if ((threadIdx.x & 31) == 0) { red[w] = s; red2[w] = s2; }
  __syncthreads();
  if (threadIdx.x == 0) {
    float S = 0.f, S2 = 0.f;
    for (int j = 0; j < 8; ++j) { S += red[j]; S2 += red2[j]; }
    float mu = S * (1.f / DM);
    float var = S2 * (1.f / DM) - mu * mu;
    bc[0] = mu; bc[1] = rsqrtf(var + 1e-5f);
  }
  __syncthreads();
  float mu = bc[0], rstd = bc[1];
  float d = 0.f;
#pragma unroll
  for (int i = 0; i < 4; ++i) {
    int idx = threadIdx.x + i * 256;
    float xv = (v[i] - mu) * rstd * g[idx] + bt[idx];
    x[(size_t)row * DM + idx] = xv;
    d += xv * clsw[idx];
  }
#pragma unroll
  for (int off = 16; off; off >>= 1) d += __shfl_xor(d, off);
  __syncthreads();
  if ((threadIdx.x & 31) == 0) red[w] = d;
  __syncthreads();
  if (threadIdx.x == 0) {
    float D = 0.f;
    for (int j = 0; j < 8; ++j) D += red[j];
    float lg = D + clsb[0];
    logits[row] = lg;
    sig[row] = 1.f / (1.f + __expf(-lg));
  }
}

extern "C" void kernel_launch(void* const* d_in, const int* in_sizes, int n_in,
                              void* d_out, int out_size, void* d_ws, size_t ws_size,
                              hipStream_t stream) {
  const float* img    = (const float*)d_in[0];
  const float* txt    = (const float*)d_in[1];
  const unsigned char* mask = (const unsigned char*)d_in[2];
  const float* in_w   = (const float*)d_in[3];
  const float* in_b   = (const float*)d_in[4];
  const float* out_w  = (const float*)d_in[5];
  const float* out_b  = (const float*)d_in[6];
  const float* ltau   = (const float*)d_in[7];
  const float* n1g    = (const float*)d_in[8];
  const float* n1b    = (const float*)d_in[9];
  const float* w1     = (const float*)d_in[10];
  const float* b1     = (const float*)d_in[11];
  const float* w2     = (const float*)d_in[12];
  const float* b2     = (const float*)d_in[13];
  const float* n2g    = (const float*)d_in[14];
  const float* n2b    = (const float*)d_in[15];
  const float* clsw   = (const float*)d_in[16];
  const float* clsb   = (const float*)d_in[17];

  float* out_x      = (float*)d_out;                       // (8,1024,1024)
  float* out_wts    = out_x + (size_t)BB * PP * DM;        // (8,16,1024,512)
  float* out_logits = out_wts + (size_t)BB * NH * PP * NN; // (8192,)
  float* out_sig    = out_logits + BB * PP;

  char* wsb = (char*)d_ws;
  size_t off = 0;
  auto carve = [&](size_t bytes) {
    void* p = wsb + off;
    off += (bytes + 255) & ~(size_t)255;
    return p;
  };
  _Float16* win16 = (_Float16*)carve((size_t)3 * DM * DM * 2);  // qkv weights f16
  _Float16* wo16  = (_Float16*)carve((size_t)DM * DM * 2);
  _Float16* w1_16 = (_Float16*)carve((size_t)HIDN * DM * 2);
  _Float16* w2_16 = (_Float16*)carve((size_t)DM * HIDN * 2);
  _Float16* q16   = (_Float16*)carve((size_t)BB * PP * DM * 2);
  _Float16* k16   = (_Float16*)carve((size_t)BB * NN * DM * 2);
  _Float16* vt16  = (_Float16*)carve((size_t)BB * NH * DH * NN * 2);
  _Float16* at16  = (_Float16*)carve((size_t)BB * PP * DM * 2);
  float*    ybuf  = (float*)carve((size_t)BB * PP * DM * 4);
  float*    x1    = (float*)carve((size_t)BB * PP * DM * 4);
  _Float16* h16   = (_Float16*)carve((size_t)BB * PP * HIDN * 2);
  (void)ws_size; (void)n_in; (void)in_sizes; (void)out_size;

  // 1) weights -> f16
  {
    int n = 3 * DM * DM;
    cvt_k<<<(n + 255) / 256, 256, 0, stream>>>(in_w, win16, n);
    n = DM * DM;
    cvt_k<<<(n + 255) / 256, 256, 0, stream>>>(out_w, wo16, n);
    n = HIDN * DM;
    cvt_k<<<(n + 255) / 256, 256, 0, stream>>>(w1, w1_16, n);
    n = DM * HIDN;
    cvt_k<<<(n + 255) / 256, 256, 0, stream>>>(w2, w2_16, n);
  }
  // 2) Q = img @ Wq^T + bq (M=8192), K = txt @ Wk^T + bk (M=4096)
  {
    int tiles = (BB * PP / 32) * (DM / 32);
    gemm_a32_f16_k<<<(tiles + 7) / 8, 256, 0, stream>>>(
        img, DM, win16, DM, in_b, q16, DM, BB * PP, DM, DM, 0);
    tiles = (BB * NN / 32) * (DM / 32);
    gemm_a32_f16_k<<<(tiles + 7) / 8, 256, 0, stream>>>(
        txt, DM, win16 + (size_t)DM * DM, DM, in_b + DM, k16, DM, BB * NN, DM, DM, 0);
    // V projection, stored transposed per head (16x16 tiles)
    int vtiles = (BB * NN / 16) * (DM / 16);
    gemm_vT_k<<<(vtiles + 7) / 8, 256, 0, stream>>>(
        txt, win16 + (size_t)2 * DM * DM, in_b + 2 * DM, vt16);
  }
  // 3) scores + softmax -> fp32 attention weights straight into d_out
  scores_softmax_k<<<BB * NH * (PP / 16), 512, 0, stream>>>(q16, k16, mask, ltau, out_wts);
  // 4) attn_out = weights @ V  (32x32 tiles: 64 per (b,h))
  gemm_attn_k<<<(BB * NH * 64) / 8, 256, 0, stream>>>(out_wts, vt16, at16);
  // 5) out-proj + residual(img)
  {
    int tiles = (BB * PP / 32) * (DM / 32);
    gemm_a16_f32res_k<<<(tiles + 7) / 8, 256, 0, stream>>>(
        at16, DM, wo16, DM, out_b, img, ybuf, DM, BB * PP, DM, DM);
  }
  // 6) LN1 -> x1
  layernorm_k<<<BB * PP, 256, 0, stream>>>(ybuf, n1g, n1b, x1);
  // 7) FFN1 (relu)
  {
    int tiles = (BB * PP / 32) * (HIDN / 32);
    gemm_a32_f16_k<<<(tiles + 7) / 8, 256, 0, stream>>>(
        x1, DM, w1_16, DM, b1, h16, HIDN, BB * PP, HIDN, DM, 1);
  }
  // 8) FFN2 + residual(x1)
  {
    int tiles = (BB * PP / 32) * (DM / 32);
    gemm_a16_f32res_k<<<(tiles + 7) / 8, 256, 0, stream>>>(
        h16, HIDN, w2_16, HIDN, b2, x1, ybuf, DM, BB * PP, DM, HIDN);
  }
  // 9) LN2 + classifier + sigmoid
  ln2_cls_k<<<BB * PP, 256, 0, stream>>>(ybuf, n2g, n2b, clsw, clsb,
                                         out_x, out_logits, out_sig);
}
